// Multihead_attention_120259084369
// MI455X (gfx1250) — matrix operand order repaired and checked by reading, not compiled, not run on hardware
//
#include <hip/hip_runtime.h>
#include <hip/hip_bf16.h>

// ---------------------------------------------------------------------------
// Multi-head attention for MI455X (gfx1250, wave32, WMMA bf16, TDM staging).
// B=4, T=2048, E=1024, H=16, HD=64.
// ---------------------------------------------------------------------------

#define B_SZ 4
#define T_SZ 2048
#define E_SZ 1024
#define H_SZ 16
#define HD_SZ 64
#define M_SZ (B_SZ * T_SZ)          // 8192
#define N_QKV (3 * E_SZ)            // 3072
#define ATT_SCALE 0.125f            // 64^-0.5

typedef __attribute__((ext_vector_type(16))) __bf16        v16bf;
typedef __attribute__((ext_vector_type(8)))  float         v8f;
typedef __attribute__((ext_vector_type(4)))  unsigned int  v4u;
typedef __attribute__((ext_vector_type(8)))  int           v8i_t;
typedef __attribute__((ext_vector_type(4)))  int           v4i_t;

#if __has_builtin(__builtin_amdgcn_tensor_load_to_lds)
#define HAVE_TDM 1
#else
#define HAVE_TDM 0
#endif
#if __has_builtin(__builtin_amdgcn_s_wait_tensorcnt)
#define WAIT_TENSOR(n) __builtin_amdgcn_s_wait_tensorcnt(n)
#else
#define WAIT_TENSOR(n)
#endif

union Frag16 {            // 16 bf16 per lane = 32 bytes = 8 VGPRs
    v16bf v;
    int4  q[2];
    __bf16 h[16];
};

__device__ __forceinline__ v8f wmma_bf16(const Frag16& a, const Frag16& b, v8f c) {
    return __builtin_amdgcn_wmma_f32_16x16x32_bf16(
        false, a.v, false, b.v, (short)0, c, false, false);
}

// ---------------------------------------------------------------------------
// TDM 2D tile load: global (row-major, stride0 elems) -> LDS with row padding.
// padIntCode: pad every 2^(c+1) DWORDs; padAmtCode: pad (a+1) DWORDs.
// bf16 elements (data_size code 1). Issue from a single wave.
// ---------------------------------------------------------------------------
__device__ __forceinline__ void tdm_load_2d(void* ldsPtr, const void* gPtr,
                                            unsigned tile0, unsigned tile1,
                                            unsigned stride0,
                                            unsigned padIntCode, unsigned padAmtCode) {
#if HAVE_TDM
    unsigned long long ga = (unsigned long long)gPtr;
    unsigned lds = (unsigned)(unsigned long long)ldsPtr;   // low 32 bits = LDS offset
    v4u g0;
    g0[0] = 1u;                                     // count=1 (valid), user mode
    g0[1] = lds;                                    // lds_addr [63:32]
    g0[2] = (unsigned)ga;                           // global_addr lo
    g0[3] = (unsigned)(ga >> 32) | 0x80000000u;     // addr hi | type=2 ([127:126]=10b)
    const unsigned td0 = 0x40000000u;               // huge tensor dims: no OOB clipping
    const unsigned td1 = 0x40000000u;
    v8i_t g1;
    g1[0] = (int)((1u << 16) |                      // data_size = 1 (2 bytes)
                  (1u << 20) |                      // pad_enable
                  (padIntCode << 22) | (padAmtCode << 25));
    g1[1] = (int)((td0 & 0xFFFFu) << 16);                       // tensor_dim0 lo16
    g1[2] = (int)((td0 >> 16) | ((td1 & 0xFFFFu) << 16));       // dim0 hi / dim1 lo
    g1[3] = (int)((td1 >> 16) | (tile0 << 16));                 // dim1 hi / tile_dim0
    g1[4] = (int)(tile1 & 0xFFFFu);                             // tile_dim1 (tile_dim2=0)
    g1[5] = (int)stride0;                                       // tensor_dim0_stride lo32
    g1[6] = 0;                                                  // stride hi / dim1_stride lo
    g1[7] = 0;
    v4i_t gz = {0, 0, 0, 0};
#if defined(__clang_major__) && __clang_major__ >= 23
    v8i_t gz8 = {0, 0, 0, 0, 0, 0, 0, 0};
    __builtin_amdgcn_tensor_load_to_lds(g0, g1, gz, gz, gz8, 0);
#else
    __builtin_amdgcn_tensor_load_to_lds(g0, g1, gz, gz, 0);
#endif
#endif
}

// Fallback cooperative copy (used only if TDM builtin is unavailable).
__device__ __forceinline__ void coop_load_2d(__bf16* dst, const __bf16* src,
                                             int rows, int rowElems, long srcStride,
                                             int dstStride, int tid, int nthr) {
    int chunks = rowElems >> 3;                     // 16B chunks per row
    for (int i = tid; i < rows * chunks; i += nthr) {
        int r = i / chunks, c = (i % chunks) * 8;
        *reinterpret_cast<int4*>(&dst[r * dstStride + c]) =
            *reinterpret_cast<const int4*>(&src[(long)r * srcStride + c]);
    }
}

// ---------------------------------------------------------------------------
// Kernel 1a: f32 -> bf16 (vectorized by 4)
// ---------------------------------------------------------------------------
__global__ void cvt_f32_bf16(const float* __restrict__ src, __bf16* __restrict__ dst, int n4) {
    int i = blockIdx.x * blockDim.x + threadIdx.x;
    if (i < n4) {
        float4 f = reinterpret_cast<const float4*>(src)[i];
        __bf16 o[4] = {(__bf16)f.x, (__bf16)f.y, (__bf16)f.z, (__bf16)f.w};
        reinterpret_cast<short4*>(dst)[i] = *reinterpret_cast<short4*>(o);
    }
}

// Kernel 1b: f32 [K][N] -> bf16 transposed [N][K] (weights; one-time)
__global__ void cvt_transpose(const float* __restrict__ src, __bf16* __restrict__ dst,
                              int N, int K) {
    int i = blockIdx.x * blockDim.x + threadIdx.x;
    if (i < N * K) {
        int n = i / K, k = i - n * K;
        dst[i] = (__bf16)src[(long)k * N + n];
    }
}

// ---------------------------------------------------------------------------
// Kernel 2: QKV GEMM  C[8192,3072] = X[8192,1024] * W + b
// A: X row-major; B: W^T [3072][1024] (K-contiguous). TDM double-buffered.
// Block 256 thr (8 waves), tile 128x128, BK=32. Wave tile 64x32.
// Epilogue: Q scaled, Q/K head-major [B*H,T,HD]; V transposed [B*H,HD,T].
// ---------------------------------------------------------------------------
#define LDA_PAD 40        // 32 + 4 DWORDs pad (pad codes: interval 16DW=3, amount 4DW=3)
#define TILE_E  (128 * LDA_PAD)

__global__ __launch_bounds__(256)
void qkv_gemm(const __bf16* __restrict__ X, const __bf16* __restrict__ WT,
              const float* __restrict__ bias,
              __bf16* __restrict__ Qo, __bf16* __restrict__ Ko, __bf16* __restrict__ Vo) {
    __shared__ __align__(16) __bf16 As[2 * TILE_E];   // [m][k]
    __shared__ __align__(16) __bf16 Bs[2 * TILE_E];   // [n][k]

    const int tid   = threadIdx.x;
    const int wave  = tid >> 5;
    const int lane  = tid & 31;
    const int l16   = lane & 15;
    const int hi    = lane >> 4;
    const int waveM = wave >> 2;   // 0..1
    const int waveN = wave & 3;    // 0..3
    const int blockM = blockIdx.y * 128;
    const int blockN = blockIdx.x * 128;

    v8f acc[4][2];
    v8f zero = {};
    for (int r = 0; r < 4; ++r) for (int c = 0; c < 2; ++c) acc[r][c] = zero;

#if HAVE_TDM
    if (wave == 0) {
        tdm_load_2d(&As[0], X  + (long)blockM * E_SZ,      32, 128, E_SZ, 3, 3);
        tdm_load_2d(&Bs[0], WT + (long)blockN * E_SZ,      32, 128, E_SZ, 3, 3);
    }
#endif

    for (int kt = 0; kt < E_SZ / 32; ++kt) {
        const int cur = kt & 1;
        const int k0  = kt * 32;
#if HAVE_TDM
        if (wave == 0) {
            if (kt + 1 < E_SZ / 32) {
                tdm_load_2d(&As[(cur ^ 1) * TILE_E], X  + (long)blockM * E_SZ + k0 + 32,
                            32, 128, E_SZ, 3, 3);
                tdm_load_2d(&Bs[(cur ^ 1) * TILE_E], WT + (long)blockN * E_SZ + k0 + 32,
                            32, 128, E_SZ, 3, 3);
                WAIT_TENSOR(2);            // current pair done (in-order)
            } else {
                WAIT_TENSOR(0);
            }
        }
#else
        coop_load_2d(&As[cur * TILE_E], X  + (long)blockM * E_SZ + k0, 128, 32, E_SZ, LDA_PAD, tid, 256);
        coop_load_2d(&Bs[cur * TILE_E], WT + (long)blockN * E_SZ + k0, 128, 32, E_SZ, LDA_PAD, tid, 256);
#endif
        __syncthreads();

        const __bf16* Ac = &As[cur * TILE_E];
        const __bf16* Bc = &Bs[cur * TILE_E];
        Frag16 af[4], bf[2];
        #pragma unroll
        for (int r = 0; r < 4; ++r) {
            const __bf16* p = &Ac[(waveM * 64 + r * 16 + l16) * LDA_PAD + hi * 8];
            af[r].q[0] = *reinterpret_cast<const int4*>(p);
            af[r].q[1] = *reinterpret_cast<const int4*>(p + 16);
        }
        #pragma unroll
        for (int c = 0; c < 2; ++c) {
            const __bf16* p = &Bc[(waveN * 32 + c * 16 + l16) * LDA_PAD + hi * 8];
            bf[c].q[0] = *reinterpret_cast<const int4*>(p);
            bf[c].q[1] = *reinterpret_cast<const int4*>(p + 16);
        }
        #pragma unroll
        for (int r = 0; r < 4; ++r)
            #pragma unroll
            for (int c = 0; c < 2; ++c)
                acc[r][c] = wmma_bf16(af[r], bf[c], acc[r][c]);
        __syncthreads();
    }

    // ---- epilogue: section is uniform per block (128 | 1024) ----
    const int sec = blockN >> 10;            // 0=Q 1=K 2=V
    const int e0  = blockN & 1023;
    #pragma unroll
    for (int r = 0; r < 4; ++r) {
        #pragma unroll
        for (int c = 0; c < 2; ++c) {
            int n  = blockN + waveN * 32 + c * 16 + l16;
            int e  = e0 + waveN * 32 + c * 16 + l16;
            int hh = e >> 6;
            int d  = e & 63;
            float bn = bias[n];
            #pragma unroll
            for (int v = 0; v < 8; ++v) {
                int m = blockM + waveM * 64 + r * 16 + v + hi * 8;
                int b = m >> 11;
                int t = m & 2047;
                float val = acc[r][c][v] + bn;
                int bh = b * H_SZ + hh;
                if (sec == 0)
                    Qo[(((long)bh * T_SZ) + t) * HD_SZ + d] = (__bf16)(val * ATT_SCALE);
                else if (sec == 1)
                    Ko[(((long)bh * T_SZ) + t) * HD_SZ + d] = (__bf16)val;
                else
                    Vo[(((long)bh * HD_SZ) + d) * T_SZ + t] = (__bf16)val;   // transposed
            }
        }
    }
}

// ---------------------------------------------------------------------------
// Kernel 3: flash attention. 128 thr (4 waves); each wave owns 16 Q rows;
// block covers 64 rows of one (b,h). S chunks of 32, TDM double-buffered K/V.
// ---------------------------------------------------------------------------
#define KS_PAD 72         // 64 + 4 DWORDs pad (codes: interval 32DW=4, amount 4DW=3)
#define VT_PAD 40         // 32 + 4 DWORDs pad (codes: 3, 3)
#define KTILE_E (32 * KS_PAD)
#define VTILE_E (64 * VT_PAD)

__global__ __launch_bounds__(128)
void flash_attn(const __bf16* __restrict__ Qb, const __bf16* __restrict__ Kb,
                const __bf16* __restrict__ Vt,
                const float* __restrict__ attn_mask,
                const unsigned char* __restrict__ kpad,
                __bf16* __restrict__ AO) {
    __shared__ __align__(16) __bf16 Ks[2 * KTILE_E];      // [s][d]
    __shared__ __align__(16) __bf16 Vs[2 * VTILE_E];      // [d][s]
    __shared__ __align__(16) __bf16 Pb[4 * 16 * VT_PAD];  // per-wave P tile [m][s]

    const int tid  = threadIdx.x;
    const int wave = tid >> 5;
    const int lane = tid & 31;
    const int l16  = lane & 15;
    const int hi   = lane >> 4;
    const int bh   = blockIdx.y;
    const int bidx = bh >> 4;
    const int hidx = bh & 15;
    const int qrow0 = blockIdx.x * 64 + wave * 16;

    const __bf16* Kbh = Kb + (long)bh * T_SZ * HD_SZ;           // [T][64]
    const __bf16* Vbh = Vt + (long)bh * HD_SZ * T_SZ;           // [64][T]

    // Q fragments (A-layout), HD=64 -> two K-steps of 32
    Frag16 qf[2];
    {
        const __bf16* qp = Qb + ((long)bh * T_SZ + qrow0 + l16) * HD_SZ;
        #pragma unroll
        for (int j = 0; j < 2; ++j) {
            int kb = j * 32 + hi * 8;
            qf[j].q[0] = *reinterpret_cast<const int4*>(qp + kb);
            qf[j].q[1] = *reinterpret_cast<const int4*>(qp + kb + 16);
        }
    }

    float mrow[8], lrow[8];
    #pragma unroll
    for (int v = 0; v < 8; ++v) { mrow[v] = -INFINITY; lrow[v] = 0.f; }
    v8f o[4];
    v8f zero = {};
    #pragma unroll
    for (int d = 0; d < 4; ++d) o[d] = zero;

#if HAVE_TDM
    if (wave == 0) {
        tdm_load_2d(&Ks[0], Kbh, 64, 32, HD_SZ, 4, 3);          // 32 rows x 64
        tdm_load_2d(&Vs[0], Vbh, 32, 64, T_SZ, 3, 3);           // 64 rows x 32
    }
#endif

    const int NS = T_SZ / 32;
    for (int st = 0; st < NS; ++st) {
        const int cur = st & 1;
        const int s0  = st * 32;
#if HAVE_TDM
        if (wave == 0) {
            if (st + 1 < NS) {
                tdm_load_2d(&Ks[(cur ^ 1) * KTILE_E], Kbh + (long)(s0 + 32) * HD_SZ,
                            64, 32, HD_SZ, 4, 3);
                tdm_load_2d(&Vs[(cur ^ 1) * VTILE_E], Vbh + s0 + 32,
                            32, 64, T_SZ, 3, 3);
                WAIT_TENSOR(2);
            } else {
                WAIT_TENSOR(0);
            }
        }
#else
        coop_load_2d(&Ks[cur * KTILE_E], Kbh + (long)s0 * HD_SZ, 32, 64, HD_SZ, KS_PAD, tid, 128);
        coop_load_2d(&Vs[cur * VTILE_E], Vbh + s0,               64, 32, T_SZ,  VT_PAD, tid, 128);
#endif
        __syncthreads();

        const __bf16* Kc = &Ks[cur * KTILE_E];
        const __bf16* Vc = &Vs[cur * VTILE_E];

        // ---- scores S[16x32] = Q(16x64) * K^T ----
        v8f s_acc[2] = {zero, zero};
        #pragma unroll
        for (int nt = 0; nt < 2; ++nt) {
            #pragma unroll
            for (int j = 0; j < 2; ++j) {
                Frag16 kf;
                const __bf16* p = &Kc[(nt * 16 + l16) * KS_PAD + j * 32 + hi * 8];
                kf.q[0] = *reinterpret_cast<const int4*>(p);
                kf.q[1] = *reinterpret_cast<const int4*>(p + 16);
                s_acc[nt] = wmma_bf16(qf[j], kf, s_acc[nt]);
            }
        }

        // ---- masks + online softmax ----
        float kp0 = kpad[bidx * T_SZ + s0 + l16]      ? -INFINITY : 0.f;
        float kp1 = kpad[bidx * T_SZ + s0 + 16 + l16] ? -INFINITY : 0.f;
        float p0[8], p1[8], alpha[8];
        #pragma unroll
        for (int v = 0; v < 8; ++v) {
            int qr = qrow0 + v + hi * 8;
            float sv0 = s_acc[0][v] + attn_mask[(long)qr * T_SZ + s0 + l16] + kp0;
            float sv1 = s_acc[1][v] + attn_mask[(long)qr * T_SZ + s0 + 16 + l16] + kp1;
            float t = fmaxf(sv0, sv1);
            t = fmaxf(t, __shfl_xor(t, 1, 16));
            t = fmaxf(t, __shfl_xor(t, 2, 16));
            t = fmaxf(t, __shfl_xor(t, 4, 16));
            t = fmaxf(t, __shfl_xor(t, 8, 16));
            float mnew = fmaxf(mrow[v], t);
            alpha[v] = __expf(mrow[v] - mnew);
            p0[v] = __expf(sv0 - mnew);
            p1[v] = __expf(sv1 - mnew);
            float rs = p0[v] + p1[v];
            rs += __shfl_xor(rs, 1, 16);
            rs += __shfl_xor(rs, 2, 16);
            rs += __shfl_xor(rs, 4, 16);
            rs += __shfl_xor(rs, 8, 16);
            lrow[v] = lrow[v] * alpha[v] + rs;
            mrow[v] = mnew;
        }
        #pragma unroll
        for (int d = 0; d < 4; ++d)
            #pragma unroll
            for (int v = 0; v < 8; ++v)
                o[d][v] *= alpha[v];

        // ---- P (C-layout f32) -> LDS -> A-fragment (bf16) ----
        __bf16* pb = &Pb[wave * 16 * VT_PAD];
        #pragma unroll
        for (int v = 0; v < 8; ++v) {
            int row = v + hi * 8;
            pb[row * VT_PAD + l16]      = (__bf16)p0[v];
            pb[row * VT_PAD + 16 + l16] = (__bf16)p1[v];
        }
        __builtin_amdgcn_wave_barrier();
        Frag16 pf;
        {
            const __bf16* p = &pb[l16 * VT_PAD + hi * 8];
            pf.q[0] = *reinterpret_cast<const int4*>(p);
            pf.q[1] = *reinterpret_cast<const int4*>(p + 16);
        }

        // ---- O(16x64) += P(16x32) * V(32x64) ----
        #pragma unroll
        for (int d = 0; d < 4; ++d) {
            Frag16 vf;
            const __bf16* p = &Vc[(d * 16 + l16) * VT_PAD + hi * 8];
            vf.q[0] = *reinterpret_cast<const int4*>(p);
            vf.q[1] = *reinterpret_cast<const int4*>(p + 16);
            o[d] = wmma_bf16(pf, vf, o[d]);
        }
        __syncthreads();
    }

    // ---- normalize + store bf16 into [B*T, E] ----
    #pragma unroll
    for (int v = 0; v < 8; ++v) lrow[v] = 1.f / lrow[v];
    #pragma unroll
    for (int d = 0; d < 4; ++d) {
        #pragma unroll
        for (int v = 0; v < 8; ++v) {
            int row = qrow0 + v + hi * 8;
            long idx = ((long)(bidx * T_SZ + row)) * E_SZ + hidx * HD_SZ + d * 16 + l16;
            AO[idx] = (__bf16)(o[d][v] * lrow[v]);
        }
    }
}

// ---------------------------------------------------------------------------
// Kernel 4: output projection  out[8192,1024] = AO * WoutT^T + b  (f32 out)
// ---------------------------------------------------------------------------
__global__ __launch_bounds__(256)
void out_gemm(const __bf16* __restrict__ X, const __bf16* __restrict__ WT,
              const float* __restrict__ bias, float* __restrict__ out) {
    __shared__ __align__(16) __bf16 As[2 * TILE_E];
    __shared__ __align__(16) __bf16 Bs[2 * TILE_E];

    const int tid   = threadIdx.x;
    const int wave  = tid >> 5;
    const int lane  = tid & 31;
    const int l16   = lane & 15;
    const int hi    = lane >> 4;
    const int waveM = wave >> 2;
    const int waveN = wave & 3;
    const int blockM = blockIdx.y * 128;
    const int blockN = blockIdx.x * 128;

    v8f acc[4][2];
    v8f zero = {};
    for (int r = 0; r < 4; ++r) for (int c = 0; c < 2; ++c) acc[r][c] = zero;

#if HAVE_TDM
    if (wave == 0) {
        tdm_load_2d(&As[0], X  + (long)blockM * E_SZ, 32, 128, E_SZ, 3, 3);
        tdm_load_2d(&Bs[0], WT + (long)blockN * E_SZ, 32, 128, E_SZ, 3, 3);
    }
#endif

    for (int kt = 0; kt < E_SZ / 32; ++kt) {
        const int cur = kt & 1;
        const int k0  = kt * 32;
#if HAVE_TDM
        if (wave == 0) {
            if (kt + 1 < E_SZ / 32) {
                tdm_load_2d(&As[(cur ^ 1) * TILE_E], X  + (long)blockM * E_SZ + k0 + 32,
                            32, 128, E_SZ, 3, 3);
                tdm_load_2d(&Bs[(cur ^ 1) * TILE_E], WT + (long)blockN * E_SZ + k0 + 32,
                            32, 128, E_SZ, 3, 3);
                WAIT_TENSOR(2);
            } else {
                WAIT_TENSOR(0);
            }
        }
#else
        coop_load_2d(&As[cur * TILE_E], X  + (long)blockM * E_SZ + k0, 128, 32, E_SZ, LDA_PAD, tid, 256);
        coop_load_2d(&Bs[cur * TILE_E], WT + (long)blockN * E_SZ + k0, 128, 32, E_SZ, LDA_PAD, tid, 256);
#endif
        __syncthreads();

        const __bf16* Ac = &As[cur * TILE_E];
        const __bf16* Bc = &Bs[cur * TILE_E];
        Frag16 af[4], bf[2];
        #pragma unroll
        for (int r = 0; r < 4; ++r) {
            const __bf16* p = &Ac[(waveM * 64 + r * 16 + l16) * LDA_PAD + hi * 8];
            af[r].q[0] = *reinterpret_cast<const int4*>(p);
            af[r].q[1] = *reinterpret_cast<const int4*>(p + 16);
        }
        #pragma unroll
        for (int c = 0; c < 2; ++c) {
            const __bf16* p = &Bc[(waveN * 32 + c * 16 + l16) * LDA_PAD + hi * 8];
            bf[c].q[0] = *reinterpret_cast<const int4*>(p);
            bf[c].q[1] = *reinterpret_cast<const int4*>(p + 16);
        }
        #pragma unroll
        for (int r = 0; r < 4; ++r)
            #pragma unroll
            for (int c = 0; c < 2; ++c)
                acc[r][c] = wmma_bf16(af[r], bf[c], acc[r][c]);
        __syncthreads();
    }

    #pragma unroll
    for (int r = 0; r < 4; ++r) {
        #pragma unroll
        for (int c = 0; c < 2; ++c) {
            int n = blockN + waveN * 32 + c * 16 + l16;
            float bn = bias[n];
            #pragma unroll
            for (int v = 0; v < 8; ++v) {
                int m = blockM + waveM * 64 + r * 16 + v + hi * 8;
                out[(long)m * E_SZ + n] = acc[r][c][v] + bn;
            }
        }
    }
}

// ---------------------------------------------------------------------------
// Host launcher
// ---------------------------------------------------------------------------
extern "C" void kernel_launch(void* const* d_in, const int* in_sizes, int n_in,
                              void* d_out, int out_size, void* d_ws, size_t ws_size,
                              hipStream_t stream) {
    const float*         q_f32  = (const float*)d_in[0];
    const unsigned char* kpad   = (const unsigned char*)d_in[1];
    const float*         amask  = (const float*)d_in[2];
    const float*         Wqkv_f = (const float*)d_in[3];
    const float*         bqkv   = (const float*)d_in[4];
    const float*         Wout_f = (const float*)d_in[5];
    const float*         bout   = (const float*)d_in[6];
    float*               out    = (float*)d_out;

    char* ws = (char*)d_ws;
    size_t off = 0;
    auto alloc = [&](size_t bytes) { void* p = ws + off; off = (off + bytes + 255) & ~size_t(255); return p; };
    __bf16* Xbf    = (__bf16*)alloc((size_t)M_SZ * E_SZ * 2);
    __bf16* WqkvT  = (__bf16*)alloc((size_t)E_SZ * N_QKV * 2);   // [3072][1024]
    __bf16* WoutT  = (__bf16*)alloc((size_t)E_SZ * E_SZ * 2);    // [1024][1024]
    __bf16* Qb     = (__bf16*)alloc((size_t)M_SZ * E_SZ * 2);    // [B*H][T][64]
    __bf16* Kb     = (__bf16*)alloc((size_t)M_SZ * E_SZ * 2);    // [B*H][T][64]
    __bf16* Vt     = (__bf16*)alloc((size_t)M_SZ * E_SZ * 2);    // [B*H][64][T]
    __bf16* AOb    = (__bf16*)alloc((size_t)M_SZ * E_SZ * 2);    // [B*T][E]
    (void)ws_size;

    {
        int n4 = (M_SZ * E_SZ) / 4;
        cvt_f32_bf16<<<(n4 + 255) / 256, 256, 0, stream>>>(q_f32, Xbf, n4);
        int n = N_QKV * E_SZ;
        cvt_transpose<<<(n + 255) / 256, 256, 0, stream>>>(Wqkv_f, WqkvT, N_QKV, E_SZ);
        n = E_SZ * E_SZ;
        cvt_transpose<<<(n + 255) / 256, 256, 0, stream>>>(Wout_f, WoutT, E_SZ, E_SZ);
    }
    {
        dim3 grid(N_QKV / 128, M_SZ / 128);     // 24 x 64
        qkv_gemm<<<grid, 256, 0, stream>>>(Xbf, WqkvT, bqkv, Qb, Kb, Vt);
    }
    {
        dim3 grid(T_SZ / 64, B_SZ * H_SZ);      // 32 x 64
        flash_attn<<<grid, 128, 0, stream>>>(Qb, Kb, Vt, amask, kpad, AOb);
    }
    {
        dim3 grid(E_SZ / 128, M_SZ / 128);      // 8 x 64
        out_gemm<<<grid, 256, 0, stream>>>(AOb, WoutT, bout, out);
    }
    (void)in_sizes; (void)n_in; (void)out_size;
}
